// ART_Restormer_79061757985198
// MI455X (gfx1250) — compile-verified, matrix operand order checked by reading, not run on hardware
//
#include <hip/hip_runtime.h>
#include <hip/hip_bf16.h>
#include <math.h>
#include <stdint.h>

typedef _Float16 half_t;
typedef __attribute__((ext_vector_type(16))) _Float16 v16h;
typedef __attribute__((ext_vector_type(8)))  _Float16 v8h;
typedef __attribute__((ext_vector_type(8)))  float    v8f;

union V16HU { v16h v; v8h h[2]; };

// Load a 16x32 f16 WMMA fragment (A, or B as B^T rows) from a K-contiguous
// [16+][ldk] half buffer. Per CDNA5 ISA 16-bit A layout: lane = m + 16*hi,
// lanes hold K = hi*8 + 0..7 (VGPR0-3) and K = 16 + hi*8 + 0..7 (VGPR4-7),
// i.e. two contiguous 16-byte spans per lane.
__device__ __forceinline__ v16h ld_frag(const half_t* base, int ldk, int lane) {
  const half_t* p = base + (lane & 15) * ldk + ((lane >> 4) << 3);
  V16HU r;
  r.h[0] = *(const v8h*)(p);
  r.h[1] = *(const v8h*)(p + 16);
  return r.v;
}

__device__ __forceinline__ v8f wmma16(v16h a, v16h b, v8f c) {
  return __builtin_amdgcn_wmma_f32_16x16x32_f16(false, a, false, b, (short)0, c,
                                                false, false);
}

// CDNA5 async global->LDS copy (ASYNCcnt-tracked, no VGPR data round-trip).
// LDS byte address = low 32 bits of the generic pointer (LDS aperture).
__device__ __forceinline__ void async_b128(half_t* lds_dst, const half_t* gsrc) {
  asm volatile("global_load_async_to_lds_b128 %0, %1, off"
               :: "v"((uint32_t)(uintptr_t)lds_dst), "v"(gsrc)
               : "memory");
}
__device__ __forceinline__ void async_wait0() {
  asm volatile("s_wait_asynccnt 0" ::: "memory");
}

// copy `chunks` 16-byte chunks global->LDS with async engine, all threads
__device__ __forceinline__ void stage_async(half_t* lds, const half_t* g,
                                            int chunks, int tid, int nthr) {
  for (int c = tid; c < chunks; c += nthr) async_b128(lds + c * 8, g + c * 8);
  async_wait0();
  __syncthreads();
}

// ---------------- weight convert: src f32 [K][N] -> dst f16 [N][K] ----------
__global__ void k_wcvt(const float* __restrict__ src, half_t* __restrict__ dst,
                       int K, int N) {
  int idx = blockIdx.x * blockDim.x + threadIdx.x;
  if (idx >= K * N) return;
  int k = idx / N, n = idx - k * N;
  dst[(size_t)n * K + k] = (half_t)src[idx];
}

// ---------------- dynamic position bias MLP ---------------------------------
__device__ __forceinline__ void ln_relu_mm(const float* x, const float* g,
                                           const float* b, const float* w,
                                           const float* bias, float* out,
                                           int nout) {
  float m = 0.f;
  for (int j = 0; j < 12; ++j) m += x[j];
  m *= (1.f / 12.f);
  float v = 0.f;
  for (int j = 0; j < 12; ++j) { float d = x[j] - m; v += d * d; }
  v *= (1.f / 12.f);
  float inv = rsqrtf(v + 1e-5f);
  float r[12];
  for (int j = 0; j < 12; ++j) {
    float y = (x[j] - m) * inv * g[j] + b[j];
    r[j] = y > 0.f ? y : 0.f;
  }
  for (int o = 0; o < nout; ++o) {
    float s = bias[o];
    for (int j = 0; j < 12; ++j) s += r[j] * w[j * nout + o];
    out[o] = s;
  }
}

__global__ void k_posbias_mlp(const float* __restrict__ pw, const float* __restrict__ pb,
                              const float* __restrict__ g1, const float* __restrict__ b1,
                              const float* __restrict__ w1, const float* __restrict__ bb1,
                              const float* __restrict__ g2, const float* __restrict__ b2,
                              const float* __restrict__ w2, const float* __restrict__ bb2,
                              const float* __restrict__ g3, const float* __restrict__ b3,
                              const float* __restrict__ w3, const float* __restrict__ bb3,
                              float* __restrict__ ptab) {
  int i = blockIdx.x * blockDim.x + threadIdx.x;
  if (i >= 961) return;
  float ri = (float)(i / 31 - 15), rj = (float)(i % 31 - 15);
  float a[12], t[12], o[6];
  for (int j = 0; j < 12; ++j) a[j] = ri * pw[j] + rj * pw[12 + j] + pb[j];
  ln_relu_mm(a, g1, b1, w1, bb1, t, 12);
  ln_relu_mm(t, g2, b2, w2, bb2, a, 12);
  ln_relu_mm(a, g3, b3, w3, bb3, o, 6);
  for (int h = 0; h < 6; ++h) ptab[i * 6 + h] = o[h];
}

__global__ void k_posbias_expand(const float* __restrict__ ptab,
                                 float* __restrict__ bias) {
  int idx = blockIdx.x * blockDim.x + threadIdx.x;  // 65536 = 256*256
  int n = idx >> 8, m = idx & 255;
  int rel0 = (n >> 4) - (m >> 4), rel1 = (n & 15) - (m & 15);
  int pi = (rel0 + 15) * 31 + (rel1 + 15);
  for (int h = 0; h < 6; ++h)
    bias[((size_t)h * 256 + n) * 256 + m] = ptab[pi * 6 + h];
}

// ---------------- layernorm (192-dim) -> f16 --------------------------------
__global__ __launch_bounds__(256) void k_layernorm(const float* __restrict__ x,
                                                   const float* __restrict__ g,
                                                   const float* __restrict__ b,
                                                   half_t* __restrict__ out) {
  const int w = threadIdx.x >> 5, lane = threadIdx.x & 31;
  const int t = blockIdx.x * 8 + w;
  const float* xr = x + (size_t)t * 192;
  float v[6];
  float s = 0.f;
  for (int i = 0; i < 6; ++i) { v[i] = xr[lane + 32 * i]; s += v[i]; }
  for (int m = 16; m; m >>= 1) s += __shfl_xor(s, m);
  const float mean = s * (1.f / 192.f);
  float q = 0.f;
  for (int i = 0; i < 6; ++i) { float d = v[i] - mean; q += d * d; }
  for (int m = 16; m; m >>= 1) q += __shfl_xor(q, m);
  const float inv = rsqrtf(q * (1.f / 192.f) + 1e-5f);
  for (int i = 0; i < 6; ++i) {
    int c = lane + 32 * i;
    out[(size_t)t * 192 + c] = (half_t)((v[i] - mean) * inv * g[c] + b[c]);
  }
}

// ---------------- QKV GEMM: [64 tokens x 192] @ [192 x 576] -----------------
__global__ __launch_bounds__(128) void k_qkv(const half_t* __restrict__ xh,
                                             const half_t* __restrict__ wh,
                                             const float* __restrict__ qkvb,
                                             half_t* __restrict__ qh,
                                             half_t* __restrict__ kh,
                                             half_t* __restrict__ vh) {
  __shared__ half_t As[64 * 192];
  const int t0 = blockIdx.x * 64;
  stage_async(As, xh + (size_t)t0 * 192, 64 * 192 / 8, threadIdx.x, 128);
  const int w = threadIdx.x >> 5, lane = threadIdx.x & 31;
  const int ln15 = lane & 15, hi = lane >> 4;
  for (int mt = 0; mt < 4; ++mt) {
    v16h aF[6];
#pragma unroll
    for (int kk = 0; kk < 6; ++kk)
      aF[kk] = ld_frag(As + mt * 16 * 192 + kk * 32, 192, lane);
    // unroll 3: ~36 b128 loads in flight per wave without blowing past 512 VGPRs
#pragma unroll 3
    for (int nt = 0; nt < 9; ++nt) {
      const int col0 = w * 144 + nt * 16;
      v16h bF[6];
#pragma unroll
      for (int kk = 0; kk < 6; ++kk)
        bF[kk] = ld_frag(wh + (size_t)col0 * 192 + kk * 32, 192, lane);
      v8f acc = {};
#pragma unroll
      for (int kk = 0; kk < 6; ++kk) acc = wmma16(aF[kk], bF[kk], acc);
      const int col = col0 + ln15;
      const int which = col / 192;
      const int head = (col % 192) >> 5;
      const int dim = col & 31;
      const float bb = qkvb[col];
      const float scale = (which == 0) ? 0.17677669529663687f : 1.0f;
      half_t* dst = (which == 0) ? qh : (which == 1) ? kh : vh;
#pragma unroll
      for (int vi = 0; vi < 8; ++vi) {
        const int t = t0 + mt * 16 + vi + (hi << 3);
        const int gy = t >> 8, gx = t & 255;
        const int win = ((gy >> 4) << 4) | (gx >> 4);
        const int n = ((gy & 15) << 4) | (gx & 15);
        dst[((size_t)(win * 6 + head) * 256 + n) * 32 + dim] =
            (half_t)((acc[vi] + bb) * scale);
      }
    }
  }
}

// ---------------- window attention: one block per (window, head) ------------
__global__ __launch_bounds__(128) void k_attn(const half_t* __restrict__ qh,
                                              const half_t* __restrict__ kh,
                                              const half_t* __restrict__ vh,
                                              const float* __restrict__ bias,
                                              half_t* __restrict__ ao) {
  __shared__ half_t Qs[256 * 32];
  __shared__ half_t Ks[256 * 32];
  __shared__ half_t Vt[32 * 256];       // V transposed: [dim][m]
  __shared__ float Ss[4][16 * 264];     // per-wave 16x256 score block (padded)
  __shared__ half_t Ps[4][16 * 264];    // per-wave probabilities (f16)
  const int win = blockIdx.x / 6, head = blockIdx.x % 6;
  const size_t base = (size_t)(win * 6 + head) * 256 * 32;
  {
    // async-stage Q and K straight into LDS
    for (int c = threadIdx.x; c < 1024; c += 128) {
      async_b128(Qs + c * 8, qh + base + c * 8);
      async_b128(Ks + c * 8, kh + base + c * 8);
    }
    // transpose V while the async engine streams Q/K
    const half_t* vsrc = vh + base;
    for (int e = threadIdx.x; e < 8192; e += 128)
      Vt[(e & 31) * 256 + (e >> 5)] = vsrc[e];
    async_wait0();
    __syncthreads();
  }
  const int w = threadIdx.x >> 5, lane = threadIdx.x & 31;
  const int ln15 = lane & 15, hi = lane >> 4;
  const float* bh = bias + (size_t)head * 65536;
  for (int i = 0; i < 4; ++i) {
    const int n0 = (w * 4 + i) * 16;
    v16h aQ = ld_frag(Qs + n0 * 32, 32, lane);
#pragma unroll 4
    for (int nt = 0; nt < 16; ++nt) {
      v16h bK = ld_frag(Ks + nt * 16 * 32, 32, lane);
      v8f acc = {};
      acc = wmma16(aQ, bK, acc);
      const int m = nt * 16 + ln15;
#pragma unroll
      for (int vi = 0; vi < 8; ++vi) {
        const int r = vi + (hi << 3);
        Ss[w][r * 264 + m] = acc[vi] + bh[(size_t)(n0 + r) * 256 + m];
      }
    }
    // row softmax over 16x256 block: 2 lanes per row (col halves), shfl-combine
    float* Sr = &Ss[w][ln15 * 264 + hi * 128];
    float mx = -3.0e38f;
    for (int c = 0; c < 128; ++c) mx = fmaxf(mx, Sr[c]);
    mx = fmaxf(mx, __shfl_xor(mx, 16));
    float sum = 0.f;
    for (int c = 0; c < 128; ++c) {
      float e = __expf(Sr[c] - mx);
      Sr[c] = e;
      sum += e;
    }
    sum += __shfl_xor(sum, 16);
    const float inv = 1.0f / sum;
    half_t* Pr = &Ps[w][ln15 * 264 + hi * 128];
    for (int c = 0; c < 128; ++c) Pr[c] = (half_t)(Sr[c] * inv);
    // O = P @ V : 16x32 output, K = 256
    for (int ot = 0; ot < 2; ++ot) {
      v8f accO = {};
#pragma unroll
      for (int kk = 0; kk < 8; ++kk) {
        v16h aP = ld_frag(Ps[w] + kk * 32, 264, lane);
        v16h bV = ld_frag(Vt + ot * 16 * 256 + kk * 32, 256, lane);
        accO = wmma16(aP, bV, accO);
      }
      const int d = ot * 16 + ln15;
#pragma unroll
      for (int vi = 0; vi < 8; ++vi) {
        const int n = n0 + vi + (hi << 3);
        ao[(size_t)(win * 256 + n) * 192 + head * 32 + d] = (half_t)accO[vi];
      }
    }
  }
}

// ---------------- proj GEMM + residual, window-reverse ----------------------
__global__ __launch_bounds__(128) void k_proj(const half_t* __restrict__ ao,
                                              const half_t* __restrict__ wh,
                                              const float* __restrict__ pb,
                                              const float* __restrict__ x,
                                              float* __restrict__ x2) {
  __shared__ half_t As[64 * 192];
  const int r0 = blockIdx.x * 64;
  stage_async(As, ao + (size_t)r0 * 192, 64 * 192 / 8, threadIdx.x, 128);
  const int w = threadIdx.x >> 5, lane = threadIdx.x & 31;
  const int ln15 = lane & 15, hi = lane >> 4;
  for (int mt = 0; mt < 4; ++mt) {
    v16h aF[6];
#pragma unroll
    for (int kk = 0; kk < 6; ++kk)
      aF[kk] = ld_frag(As + mt * 16 * 192 + kk * 32, 192, lane);
#pragma unroll
    for (int nt3 = 0; nt3 < 3; ++nt3) {
      const int col0 = (w * 3 + nt3) * 16;
      v16h bF[6];
#pragma unroll
      for (int kk = 0; kk < 6; ++kk)
        bF[kk] = ld_frag(wh + (size_t)col0 * 192 + kk * 32, 192, lane);
      v8f acc = {};
#pragma unroll
      for (int kk = 0; kk < 6; ++kk) acc = wmma16(aF[kk], bF[kk], acc);
      const int col = col0 + ln15;
      const float bb = pb[col];
#pragma unroll
      for (int vi = 0; vi < 8; ++vi) {
        const int r = r0 + mt * 16 + vi + (hi << 3);
        const int win = r >> 8, n = r & 255;
        const int gy = ((win >> 4) << 4) | (n >> 4);
        const int gx = ((win & 15) << 4) | (n & 15);
        const size_t t = (size_t)((gy << 8) | gx);
        x2[t * 192 + col] = x[t * 192 + col] + acc[vi] + bb;
      }
    }
  }
}

// ---------------- FC1 GEMM + exact GELU -------------------------------------
__global__ __launch_bounds__(128) void k_fc1(const half_t* __restrict__ yh,
                                             const half_t* __restrict__ wh,
                                             const float* __restrict__ fb,
                                             half_t* __restrict__ ho) {
  __shared__ half_t As[64 * 192];
  const int t0 = blockIdx.x * 64;
  stage_async(As, yh + (size_t)t0 * 192, 64 * 192 / 8, threadIdx.x, 128);
  const int w = threadIdx.x >> 5, lane = threadIdx.x & 31;
  const int ln15 = lane & 15, hi = lane >> 4;
  for (int mt = 0; mt < 4; ++mt) {
    v16h aF[6];
#pragma unroll
    for (int kk = 0; kk < 6; ++kk)
      aF[kk] = ld_frag(As + mt * 16 * 192 + kk * 32, 192, lane);
#pragma unroll 2
    for (int j = 0; j < 12; ++j) {
      const int col0 = (w * 12 + j) * 16;
      v16h bF[6];
#pragma unroll
      for (int kk = 0; kk < 6; ++kk)
        bF[kk] = ld_frag(wh + (size_t)col0 * 192 + kk * 32, 192, lane);
      v8f acc = {};
#pragma unroll
      for (int kk = 0; kk < 6; ++kk) acc = wmma16(aF[kk], bF[kk], acc);
      const int col = col0 + ln15;
      const float bb = fb[col];
#pragma unroll
      for (int vi = 0; vi < 8; ++vi) {
        const int t = t0 + mt * 16 + vi + (hi << 3);
        const float u = acc[vi] + bb;
        const float gel = 0.5f * u * (1.0f + erff(u * 0.70710678118654752f));
        ho[(size_t)t * 768 + col] = (half_t)gel;
      }
    }
  }
}

// ---------------- FC2 GEMM + residual -> d_out ------------------------------
__global__ __launch_bounds__(128) void k_fc2(const half_t* __restrict__ hh,
                                             const half_t* __restrict__ wh,
                                             const float* __restrict__ fb,
                                             const float* __restrict__ x2,
                                             float* __restrict__ out) {
  __shared__ half_t As[64 * 768];  // 96 KB
  const int t0 = blockIdx.x * 64;
  stage_async(As, hh + (size_t)t0 * 768, 64 * 768 / 8, threadIdx.x, 128);
  const int w = threadIdx.x >> 5, lane = threadIdx.x & 31;
  const int ln15 = lane & 15, hi = lane >> 4;
  for (int mt = 0; mt < 4; ++mt) {
    for (int nt3 = 0; nt3 < 3; ++nt3) {
      const int col0 = (w * 3 + nt3) * 16;
      const half_t* aBase = As + mt * 16 * 768;
      const half_t* bBase = wh + (size_t)col0 * 768;
      // software-pipelined K loop: load K-step kk+1 while WMMA kk executes
      v16h aC = ld_frag(aBase, 768, lane);
      v16h bC = ld_frag(bBase, 768, lane);
      v8f acc = {};
#pragma unroll
      for (int kk = 0; kk < 24; ++kk) {
        const int nx = (kk == 23) ? 0 : kk + 1;
        v16h aN = ld_frag(aBase + nx * 32, 768, lane);
        v16h bN = ld_frag(bBase + nx * 32, 768, lane);
        acc = wmma16(aC, bC, acc);
        aC = aN;
        bC = bN;
      }
      const int col = col0 + ln15;
      const float bb = fb[col];
#pragma unroll
      for (int vi = 0; vi < 8; ++vi) {
        const size_t t = (size_t)(t0 + mt * 16 + vi + (hi << 3));
        out[t * 192 + col] = x2[t * 192 + col] + acc[vi] + bb;
      }
    }
  }
}

extern "C" void kernel_launch(void* const* d_in, const int* in_sizes, int n_in,
                              void* d_out, int out_size, void* d_ws,
                              size_t ws_size, hipStream_t stream) {
  (void)in_sizes; (void)n_in; (void)out_size; (void)ws_size;
  const float* x = (const float*)d_in[0];
  const float* gamma1 = (const float*)d_in[1];
  const float* beta1 = (const float*)d_in[2];
  const float* qkv_w = (const float*)d_in[3];
  const float* qkv_b = (const float*)d_in[4];
  const float* proj_w = (const float*)d_in[5];
  const float* proj_b = (const float*)d_in[6];
  const float* pos_proj_w = (const float*)d_in[7];
  const float* pos_proj_b = (const float*)d_in[8];
  const float* ln1_g = (const float*)d_in[9];
  const float* ln1_b = (const float*)d_in[10];
  const float* pos1_w = (const float*)d_in[11];
  const float* pos1_b = (const float*)d_in[12];
  const float* ln2_g = (const float*)d_in[13];
  const float* ln2_b = (const float*)d_in[14];
  const float* pos2_w = (const float*)d_in[15];
  const float* pos2_b = (const float*)d_in[16];
  const float* ln3_g = (const float*)d_in[17];
  const float* ln3_b = (const float*)d_in[18];
  const float* pos3_w = (const float*)d_in[19];
  const float* pos3_b = (const float*)d_in[20];
  const float* gamma2 = (const float*)d_in[21];
  const float* beta2 = (const float*)d_in[22];
  const float* fc1_w = (const float*)d_in[23];
  const float* fc1_b = (const float*)d_in[24];
  const float* fc2_w = (const float*)d_in[25];
  const float* fc2_b = (const float*)d_in[26];

  char* ws = (char*)d_ws;
  size_t off = 0;
  auto alloc = [&](size_t bytes) -> char* {
    char* p = ws + off;
    off += (bytes + 255) & ~(size_t)255;
    return p;
  };
  const size_t L = 65536;
  float* ptab = (float*)alloc(961 * 6 * 4);
  float* bias = (float*)alloc((size_t)6 * 65536 * 4);
  half_t* qkv_wh = (half_t*)alloc((size_t)576 * 192 * 2);
  half_t* proj_wh = (half_t*)alloc((size_t)192 * 192 * 2);
  half_t* fc1_wh = (half_t*)alloc((size_t)768 * 192 * 2);
  half_t* fc2_wh = (half_t*)alloc((size_t)192 * 768 * 2);
  float* x2 = (float*)alloc(L * 192 * 4);
  half_t* regionA = (half_t*)alloc((size_t)4 * L * 192 * 2);  // xh,q,k,v -> h
  half_t* regionB = (half_t*)alloc(L * 192 * 2);              // ao -> yh
  half_t* xh = regionA;
  half_t* qh = regionA + L * 192;
  half_t* kh = qh + L * 192;
  half_t* vh = kh + L * 192;
  half_t* hbuf = regionA;  // fc1 output [L][768], reuses xh/q/k/v (same bytes)
  half_t* ao = regionB;
  half_t* yh = regionB;    // reuses ao after proj consumed it

  k_wcvt<<<(192 * 576 + 255) / 256, 256, 0, stream>>>(qkv_w, qkv_wh, 192, 576);
  k_wcvt<<<(192 * 192 + 255) / 256, 256, 0, stream>>>(proj_w, proj_wh, 192, 192);
  k_wcvt<<<(192 * 768 + 255) / 256, 256, 0, stream>>>(fc1_w, fc1_wh, 192, 768);
  k_wcvt<<<(768 * 192 + 255) / 256, 256, 0, stream>>>(fc2_w, fc2_wh, 768, 192);
  k_posbias_mlp<<<4, 256, 0, stream>>>(pos_proj_w, pos_proj_b, ln1_g, ln1_b,
                                       pos1_w, pos1_b, ln2_g, ln2_b, pos2_w,
                                       pos2_b, ln3_g, ln3_b, pos3_w, pos3_b,
                                       ptab);
  k_posbias_expand<<<256, 256, 0, stream>>>(ptab, bias);
  k_layernorm<<<8192, 256, 0, stream>>>(x, gamma1, beta1, xh);
  k_qkv<<<1024, 128, 0, stream>>>(xh, qkv_wh, qkv_b, qh, kh, vh);
  k_attn<<<1536, 128, 0, stream>>>(qh, kh, vh, bias, ao);
  k_proj<<<1024, 128, 0, stream>>>(ao, proj_wh, proj_b, x, x2);
  k_layernorm<<<8192, 256, 0, stream>>>(x2, gamma2, beta2, yh);
  k_fc1<<<1024, 128, 0, stream>>>(yh, fc1_wh, fc1_b, hbuf);
  k_fc2<<<1024, 128, 0, stream>>>(hbuf, fc2_wh, fc2_b, x2, (float*)d_out);
}
